// LiquidRON_44435731644538
// MI455X (gfx1250) — matrix operand order repaired and checked by reading, not compiled, not run on hardware
//
#include <hip/hip_runtime.h>

typedef __attribute__((ext_vector_type(16))) __bf16 v16bf;
typedef __attribute__((ext_vector_type(8)))  __bf16 v8bf;
typedef __attribute__((ext_vector_type(8)))  float  v8f;

#define B_    128
#define T_    1000
#define NIN_  128
#define NH_   512
#define ROWSTRIDE 520   // halfs: 1040 B = 65*16 B -> b128-aligned rows, bank offset 4

// ---------------------------------------------------------------------------
// Pack weights (fp32 row-major [K,512]) into bf16 WMMA B-fragment order.
// Fragment layout per (ntile,ktile): lane holds 16 contiguous bf16:
//   n = ntile*16 + (lane&15),  k = ktile*32 + (lane>>4)*8 + (j<8 ? j : j+8)   (j = 0..15)
// ---------------------------------------------------------------------------
__global__ __launch_bounds__(256) void pack_weights(const float* __restrict__ h2h,
                                                    const float* __restrict__ x2h,
                                                    __bf16* __restrict__ hp,
                                                    __bf16* __restrict__ xp) {
  int idx = blockIdx.x * 256 + threadIdx.x;
  if (idx < 32 * 16 * 32 * 16) {            // h2h: 32 ntiles x 16 ktiles
    int j  = idx & 15;
    int ln = (idx >> 4) & 31;
    int kt = (idx >> 9) & 15;
    int nt = idx >> 13;
    int n = nt * 16 + (ln & 15);
    int k = kt * 32 + (ln >> 4) * 8 + (j < 8 ? j : j + 8);
    hp[idx] = (__bf16)h2h[k * NH_ + n];
  }
  if (idx < 32 * 4 * 32 * 16) {             // x2h: 32 ntiles x 4 ktiles
    int j  = idx & 15;
    int ln = (idx >> 4) & 31;
    int kt = (idx >> 9) & 3;
    int nt = idx >> 11;
    int n = nt * 16 + (ln & 15);
    int k = kt * 32 + (ln >> 4) * 8 + (j < 8 ? j : j + 8);
    xp[idx] = (__bf16)x2h[k * NH_ + n];
  }
}

// ---------------------------------------------------------------------------
// Fully parallel pre-GEMM: xw[b*T+t, n] = sum_k x[b,t,k] * x2h[k,n]
// Written into the u_seq region of d_out; consumed (and overwritten) by the
// recurrence kernel as the per-step accumulator init.
// One wave per 16x16 output tile; K = 128 -> 4 bf16 WMMAs.
// ---------------------------------------------------------------------------
__global__ __launch_bounds__(256) void xw_gemm(const float* __restrict__ x,
                                               const __bf16* __restrict__ xp,
                                               float* __restrict__ out_u) {
  int wid  = blockIdx.x * 8 + (threadIdx.x >> 5);
  int lane = threadIdx.x & 31;
  int rt = wid >> 5;            // row tile: 0..7999  (B*T/16)
  int nt = wid & 31;            // col tile: 0..31
  int m  = lane & 15;
  int hi = lane >> 4;
  const float* xr = x + (size_t)(rt * 16 + m) * NIN_;
  const v16bf* bp = (const v16bf*)xp;
  v8f acc = {0.f, 0.f, 0.f, 0.f, 0.f, 0.f, 0.f, 0.f};
#pragma unroll
  for (int kt = 0; kt < 4; ++kt) {
    // A fragment (16-bit A 16x32 layout): k = kt*32 + hi*8 + (j<8 ? j : j+8)
    const float* p = xr + kt * 32 + hi * 8;
    float4 f0 = *(const float4*)(p);
    float4 f1 = *(const float4*)(p + 4);
    float4 f2 = *(const float4*)(p + 16);
    float4 f3 = *(const float4*)(p + 20);
    v16bf a;
    a[0]  = (__bf16)f0.x; a[1]  = (__bf16)f0.y; a[2]  = (__bf16)f0.z; a[3]  = (__bf16)f0.w;
    a[4]  = (__bf16)f1.x; a[5]  = (__bf16)f1.y; a[6]  = (__bf16)f1.z; a[7]  = (__bf16)f1.w;
    a[8]  = (__bf16)f2.x; a[9]  = (__bf16)f2.y; a[10] = (__bf16)f2.z; a[11] = (__bf16)f2.w;
    a[12] = (__bf16)f3.x; a[13] = (__bf16)f3.y; a[14] = (__bf16)f3.z; a[15] = (__bf16)f3.w;
    v16bf b = bp[(nt * 4 + kt) * 32 + lane];
    acc = __builtin_amdgcn_wmma_f32_16x16x32_bf16(false, a, false, b, (short)0, acc,
                                                  false, false);
  }
  int col = nt * 16 + m;
#pragma unroll
  for (int r = 0; r < 8; ++r) {
    int row = rt * 16 + r + 8 * hi;         // C/D layout: VGPR r -> M=r / M=r+8
    out_u[(size_t)row * NH_ + col] = acc[r];
  }
}

// ---------------------------------------------------------------------------
// Persistent recurrence: 8 workgroups (one per 16-row batch tile), 8 waves.
// Wave w owns hidden columns [64w, 64w+64) -> 4 N-tiles; h2h B-fragments for
// those columns are streamed from L2 each step; u (bf16) is shared in LDS.
// ---------------------------------------------------------------------------
__global__ __launch_bounds__(256) void liquid_recur(const __bf16* __restrict__ hp,
                                                    float* __restrict__ out_u,
                                                    float* __restrict__ out_s) {
  __shared__ alignas(16) __bf16 ubuf[16][ROWSTRIDE];
  const int lane = threadIdx.x & 31;
  const int w    = threadIdx.x >> 5;   // wave 0..7
  const int bt   = blockIdx.x;         // batch tile 0..7
  const int m    = lane & 15;
  const int hi   = lane >> 4;

  for (int i = threadIdx.x; i < 16 * ROWSTRIDE; i += 256)
    (&ubuf[0][0])[i] = (__bf16)0.f;    // u0 = 0

  v8f ust[4];                          // fp32 u state for this wave's 4 tiles
#pragma unroll
  for (int nt = 0; nt < 4; ++nt)
#pragma unroll
    for (int r = 0; r < 8; ++r) ust[nt][r] = 0.f;

  const v16bf* bp = (const v16bf*)hp;
  __syncthreads();

  for (int t = 0; t < T_; ++t) {
    // --- capture full 16x512 A (u, bf16) into 128 VGPRs; reused for 4 N-tiles
    v16bf afr[16];
#pragma unroll
    for (int kt = 0; kt < 16; ++kt) {
      const __bf16* p = &ubuf[m][kt * 32 + hi * 8];
      v8bf lo = *(const v8bf*)(p);        // k = kt*32 + hi*8 + 0..7
      v8bf hj = *(const v8bf*)(p + 16);   // k = kt*32 + 16 + hi*8 + 0..7
#pragma unroll
      for (int j = 0; j < 8; ++j) { afr[kt][j] = lo[j]; afr[kt][8 + j] = hj[j]; }
    }
    __syncthreads();                      // A captured -> ubuf may be rewritten

#pragma unroll
    for (int nt = 0; nt < 4; ++nt) {
      const int ntile = w * 4 + nt;
      const int col   = ntile * 16 + m;
      v8f acc;
#pragma unroll
      for (int r = 0; r < 8; ++r) {       // init with precomputed x@x2h
        const int brow = bt * 16 + r + 8 * hi;
        acc[r] = out_u[((size_t)brow * T_ + t) * NH_ + col];
      }
#pragma unroll
      for (int kt = 0; kt < 16; ++kt) {   // u @ h2h, K = 512
        v16bf b = bp[(ntile * 16 + kt) * 32 + lane];
        acc = __builtin_amdgcn_wmma_f32_16x16x32_bf16(false, afr[kt], false, b,
                                                      (short)0, acc, false, false);
      }
#pragma unroll
      for (int r = 0; r < 8; ++r) {       // spike / leak / reset, emit outputs
        const int brow = bt * 16 + r + 8 * hi;
        const size_t o = ((size_t)brow * T_ + t) * NH_ + col;
        float uo = ust[nt][r];
        float sp = (uo > 1.0f) ? 1.0f : 0.0f;
        float un = 0.98f * uo + 0.02f * acc[r];
        un = (sp != 0.0f) ? 0.0f : un;
        out_u[o] = un;                    // overwrites the xw value read above
        out_s[o] = sp;
        ust[nt][r] = un;
        ubuf[r + 8 * hi][col] = (__bf16)un;
      }
    }
    __syncthreads();                      // next step's A visible to all waves
  }
}

// ---------------------------------------------------------------------------
extern "C" void kernel_launch(void* const* d_in, const int* in_sizes, int n_in,
                              void* d_out, int out_size, void* d_ws, size_t ws_size,
                              hipStream_t stream) {
  const float* x   = (const float*)d_in[0];   // [B, T, NIN]
  const float* h2h = (const float*)d_in[1];   // [512, 512]
  const float* x2h = (const float*)d_in[2];   // [128, 512]
  float* out_u = (float*)d_out;                                  // [B, T, H]
  float* out_s = out_u + (size_t)B_ * T_ * NH_;                  // [B, T, H]
  __bf16* hp = (__bf16*)d_ws;                 // 262144 bf16 (512 KB)
  __bf16* xp = hp + 32 * 16 * 32 * 16;        //  65536 bf16 (128 KB)

  pack_weights<<<dim3(1024), dim3(256), 0, stream>>>(h2h, x2h, hp, xp);
  xw_gemm<<<dim3((B_ * T_ / 16) * 32 / 8), dim3(256), 0, stream>>>(x, xp, out_u);
  liquid_recur<<<dim3(8), dim3(256), 0, stream>>>(hp, out_u, out_s);
}
